// MKAForGPT2Attention_10548439679391
// MI455X (gfx1250) — matrix operand order repaired and checked by reading, not compiled, not run on hardware
//
#include <hip/hip_runtime.h>

// ---------------------------------------------------------------------------
// Types for CDNA5 WMMA (wave32): v_wmma_f32_16x16x32_bf16
// ---------------------------------------------------------------------------
typedef __bf16 bf16_t;
typedef __attribute__((ext_vector_type(8)))  bf16_t v8bf;
typedef __attribute__((ext_vector_type(16))) bf16_t v16bf;
typedef __attribute__((ext_vector_type(8)))  float  v8f;

// Problem constants (B=2, T=2048, C=1024, H=16, D=64)
#define BATCH 2
#define SEQ   2048
#define CH    1024
#define HEADS 16
#define HD    64
#define ROWS  (BATCH * SEQ)     // 4096 GEMM rows
#define QSCALE 0.125f           // D^-0.5

__device__ __forceinline__ v8f wmma_bf16(v16bf a, v16bf b, v8f c) {
  return __builtin_amdgcn_wmma_f32_16x16x32_bf16(
      /*neg_a=*/false, a, /*neg_b=*/false, b,
      /*c_mod=*/(short)0, c, /*reuse_a=*/false, /*reuse_b=*/false);
}

// A-fragment (16x32 bf16, MxK): lane<16 holds row M=lane, K = {0..15};
// lane>=16 holds row M=lane-16, K = {16..31}. Two 16B contiguous loads.
__device__ __forceinline__ v16bf load_a_frag(const bf16_t* rowptr, int kbase, int half) {
  v8bf lo = *(const v8bf*)(rowptr + kbase + half * 16);
  v8bf hi = *(const v8bf*)(rowptr + kbase + half * 16 + 8);
  return __builtin_shufflevector(lo, hi, 0, 1, 2, 3, 4, 5, 6, 7,
                                         8, 9, 10, 11, 12, 13, 14, 15);
}

// CDNA5 async copy: LDS[ldsaddr] = MEM[gaddr], 16B per lane, tracked by ASYNCcnt.
__device__ __forceinline__ void async_b128(unsigned int ldsaddr, unsigned long long gaddr) {
  asm volatile("global_load_async_to_lds_b128 %0, %1, off"
               :: "v"(ldsaddr), "v"(gaddr) : "memory");
}

// ---------------------------------------------------------------------------
// Prep: fp32 -> bf16 convert, and weight transpose-convert ([in][out]->[out][in])
// ---------------------------------------------------------------------------
__global__ void __launch_bounds__(256) cvt_bf16_kernel(const float* __restrict__ src,
                                                       bf16_t* __restrict__ dst, int n) {
  int i = blockIdx.x * blockDim.x + threadIdx.x;
  if (i < n) dst[i] = (bf16_t)src[i];
}

__global__ void __launch_bounds__(256) transpose_bf16_kernel(const float* __restrict__ src,
                                                             bf16_t* __restrict__ dst) {
  __shared__ float tile[16][17];
  int n = blockIdx.x * 16 + threadIdx.x;   // output-channel
  int k = blockIdx.y * 16 + threadIdx.y;   // input-channel
  tile[threadIdx.y][threadIdx.x] = src[(size_t)k * CH + n];
  __syncthreads();
  int on = blockIdx.x * 16 + threadIdx.y;
  int ok = blockIdx.y * 16 + threadIdx.x;
  dst[(size_t)on * CH + ok] = (bf16_t)tile[threadIdx.x][threadIdx.y];
}

// ---------------------------------------------------------------------------
// Fused QKV projection GEMM: [4096x1024] x [1024x1024] (x3)
// One wave computes a 16x64 output tile (64 cols == one head for Q/K/V).
// Q is pre-scaled by D^-0.5; V is stored transposed Vt[b][h][d][t].
// ---------------------------------------------------------------------------
__global__ void __launch_bounds__(256) qkv_gemm_kernel(
    const bf16_t* __restrict__ Xbf,
    const bf16_t* __restrict__ Wqt, const bf16_t* __restrict__ Wkt,
    const bf16_t* __restrict__ Wvt,
    const float* __restrict__ bq, const float* __restrict__ bk,
    const float* __restrict__ bv,
    bf16_t* __restrict__ Qbf, bf16_t* __restrict__ Kbf, bf16_t* __restrict__ Vt) {
  const int lane = threadIdx.x & 31;
  const int wave = threadIdx.x >> 5;
  const int half = lane >> 4;
  const int lcol = lane & 15;

  int tile = blockIdx.x * 8 + wave;                 // 0 .. 3*256*16-1
  const int which = tile / ((ROWS / 16) * (CH / 64)); // 0=Q 1=K 2=V
  tile -= which * (ROWS / 16) * (CH / 64);
  const int mrow = tile / (CH / 64);                // 16-row tile index
  const int nt   = tile % (CH / 64);                // 64-col tile == head id

  const bf16_t* W    = (which == 0) ? Wqt : (which == 1) ? Wkt : Wvt;
  const float*  bias = (which == 0) ? bq  : (which == 1) ? bk  : bv;

  const bf16_t* arowp = Xbf + (size_t)(mrow * 16 + lcol) * CH;

  v8f acc[4] = {{}, {}, {}, {}};
  for (int k0 = 0; k0 < CH; k0 += 32) {
    v16bf a = load_a_frag(arowp, k0, half);
#pragma unroll
    for (int j = 0; j < 4; ++j) {
      const int n = nt * 64 + j * 16 + lcol;        // output channel (B-frag col)
      v16bf b = *(const v16bf*)(W + (size_t)n * CH + k0 + half * 16);
      acc[j] = wmma_bf16(a, b, acc[j]);
    }
  }

  const float scale = (which == 0) ? QSCALE : 1.0f;
#pragma unroll
  for (int j = 0; j < 4; ++j) {
    const int n = nt * 64 + j * 16 + lcol;
    const int d = j * 16 + lcol;                    // n % 64
    const float bval = bias[n];
#pragma unroll
    for (int r = 0; r < 8; ++r) {
      const int g  = mrow * 16 + half * 8 + r;      // row in [0, 4096)
      const int bb = g >> 11;                       // batch
      const int tt = g & (SEQ - 1);                 // token
      const bf16_t ov = (bf16_t)((acc[j][r] + bval) * scale);
      if (which == 2) {
        Vt[(((size_t)bb * HEADS + nt) * HD + d) * SEQ + tt] = ov;
      } else if (which == 0) {
        Qbf[(((size_t)bb * HEADS + nt) * SEQ + tt) * HD + d] = ov;
      } else {
        Kbf[(((size_t)bb * HEADS + nt) * SEQ + tt) * HD + d] = ov;
      }
    }
  }
}

// ---------------------------------------------------------------------------
// Flash-attention: one wave per (b, h, 16-query tile). Online softmax, scores
// never leave registers. K/V blocks are double-buffered into LDS with
// global_load_async_to_lds_b128 (ASYNCcnt) so the next block's copy overlaps
// the current block's 8 WMMAs + softmax.
// ---------------------------------------------------------------------------
#define LDSTRIDE 40   // P-tile: 80B row stride, 16B-aligned, conflict-free
#define KROWB 144     // K stage: 128B row + 16B pad (16 distinct banks / half)
#define VROWB 80      // V stage: 64B row + 16B pad

__global__ void __launch_bounds__(32) attn_kernel(
    const bf16_t* __restrict__ Qbf, const bf16_t* __restrict__ Kbf,
    const bf16_t* __restrict__ Vt, bf16_t* __restrict__ Ctx) {
  __shared__ __align__(16) unsigned char kst[2][32 * KROWB]; // 2 x 4.5 KB
  __shared__ __align__(16) unsigned char vst[2][64 * VROWB]; // 2 x 5.0 KB
  __shared__ __align__(16) bf16_t plds[16 * LDSTRIDE];       // 1.25 KB

  const int lane = threadIdx.x & 31;
  const int half = lane >> 4;
  const int lcol = lane & 15;
  const int qt = blockIdx.x;       // 0..127
  const int h  = blockIdx.y;       // 0..15
  const int b  = blockIdx.z;       // 0..1
  const size_t bh = (size_t)b * HEADS + h;

  // Q A-fragments for d=0..31 and d=32..63, kept in registers for entire loop
  const bf16_t* qrow = Qbf + (bh * SEQ + qt * 16 + lcol) * HD;
  const v16bf aq0 = load_a_frag(qrow, 0, half);
  const v16bf aq1 = load_a_frag(qrow, 32, half);

  // Async-stage one 32-key block (K: 32x64 bf16, V: 64 d-rows x 32 keys)
  auto issue_block = [&](int kb, int buf) {
    // K block: 4 KB contiguous in Kbf; lane copies its token row (128B = 8x16B)
    const unsigned long long kg =
        (unsigned long long)(size_t)(Kbf + (bh * SEQ + kb * 32 + lane) * HD);
    const unsigned int kdst = (unsigned int)(size_t)&kst[buf][lane * KROWB];
#pragma unroll
    for (int c = 0; c < 8; ++c) async_b128(kdst + c * 16, kg + c * 16);
    // V block: 64 rows of 64B (stride SEQ*2B); lane copies 2 d-rows (4x16B each)
#pragma unroll
    for (int i = 0; i < 2; ++i) {
      const int d = lane * 2 + i;
      const unsigned long long vg =
          (unsigned long long)(size_t)(Vt + (bh * HD + d) * SEQ + kb * 32);
      const unsigned int vdst = (unsigned int)(size_t)&vst[buf][d * VROWB];
#pragma unroll
      for (int c = 0; c < 4; ++c) async_b128(vdst + c * 16, vg + c * 16);
    }
  };

  v8f o0 = {}, o1 = {}, o2 = {}, o3 = {};
  float m[8], l[8];
#pragma unroll
  for (int r = 0; r < 8; ++r) { m[r] = -3.0e38f; l[r] = 0.0f; }

  const int kbmax = (qt * 16 + 15) >> 5;            // inclusive causal bound
  issue_block(0, 0);                                // prologue fill

  for (int kb = 0; kb <= kbmax; ++kb) {
    const int cur = kb & 1;
    const bool has_next = (kb < kbmax);
    if (has_next) issue_block(kb + 1, cur ^ 1);     // overlap next copy

    // Wait for current buffer: 16 async issues remain in flight for next block.
    if (has_next) asm volatile("s_wait_asynccnt 0x10" ::: "memory");
    else          asm volatile("s_wait_asynccnt 0x0"  ::: "memory");

    const int kbase = kb * 32;

    // ---- S = Q K^T for two 16-key tiles (scale pre-folded into Q) ----
    v8f s[2];
#pragma unroll
    for (int t = 0; t < 2; ++t) {
      const unsigned char* kp = &kst[cur][(t * 16 + lcol) * KROWB + half * 32];
      v16bf bk0 = *(const v16bf*)(kp);
      v16bf bk1 = *(const v16bf*)(kp + 64);
      v8f c = {};
      c = wmma_bf16(aq0, bk0, c);
      c = wmma_bf16(aq1, bk1, c);
      s[t] = c;
    }

    // ---- causal mask (C-layout: col = lcol of tile, row M = half*8 + r) ----
#pragma unroll
    for (int t = 0; t < 2; ++t) {
      const int col = kbase + t * 16 + lcol;
#pragma unroll
      for (int r = 0; r < 8; ++r) {
        const int row = qt * 16 + half * 8 + r;
        if (col > row) s[t][r] = -3.0e38f;
      }
    }

    // ---- online softmax: row reductions across the 16 lanes of each half ----
#pragma unroll
    for (int r = 0; r < 8; ++r) {
      float mx = fmaxf(s[0][r], s[1][r]);
      mx = fmaxf(mx, __shfl_xor(mx, 1, 32));
      mx = fmaxf(mx, __shfl_xor(mx, 2, 32));
      mx = fmaxf(mx, __shfl_xor(mx, 4, 32));
      mx = fmaxf(mx, __shfl_xor(mx, 8, 32));
      const float mn  = fmaxf(m[r], mx);
      const float fac = __expf(m[r] - mn);
      m[r] = mn;
      const float p0 = __expf(s[0][r] - mn);
      const float p1 = __expf(s[1][r] - mn);
      s[0][r] = p0;
      s[1][r] = p1;
      float ps = p0 + p1;
      ps += __shfl_xor(ps, 1, 32);
      ps += __shfl_xor(ps, 2, 32);
      ps += __shfl_xor(ps, 4, 32);
      ps += __shfl_xor(ps, 8, 32);
      l[r] = l[r] * fac + ps;
      o0[r] *= fac; o1[r] *= fac; o2[r] *= fac; o3[r] *= fac;
    }

    // ---- reshape P: C-layout -> A-layout via padded LDS tile ----
#pragma unroll
    for (int t = 0; t < 2; ++t)
#pragma unroll
      for (int r = 0; r < 8; ++r)
        plds[(half * 8 + r) * LDSTRIDE + t * 16 + lcol] = (bf16_t)s[t][r];
    __syncthreads();
    v16bf ap;
    {
      const bf16_t* lp = plds + lcol * LDSTRIDE;
      v8bf lo = *(const v8bf*)(lp + half * 16);
      v8bf hi = *(const v8bf*)(lp + half * 16 + 8);
      ap = __builtin_shufflevector(lo, hi, 0, 1, 2, 3, 4, 5, 6, 7,
                                           8, 9, 10, 11, 12, 13, 14, 15);
    }
    __syncthreads();

    // ---- O += P * V from the staged Vt tile ----
#pragma unroll
    for (int dn = 0; dn < 4; ++dn) {
      const unsigned char* vp = &vst[cur][(dn * 16 + lcol) * VROWB + half * 32];
      v16bf bvf = *(const v16bf*)vp;
      if (dn == 0)      o0 = wmma_bf16(ap, bvf, o0);
      else if (dn == 1) o1 = wmma_bf16(ap, bvf, o1);
      else if (dn == 2) o2 = wmma_bf16(ap, bvf, o2);
      else              o3 = wmma_bf16(ap, bvf, o3);
    }
  }

  // ---- normalize, store ctx row-major [4096][1024] bf16 for the O-proj ----
#pragma unroll
  for (int r = 0; r < 8; ++r) {
    const float inv = 1.0f / l[r];
    const size_t g = (size_t)b * SEQ + qt * 16 + half * 8 + r;
    bf16_t* crow = Ctx + g * CH + h * HD + lcol;
    crow[0]  = (bf16_t)(o0[r] * inv);
    crow[16] = (bf16_t)(o1[r] * inv);
    crow[32] = (bf16_t)(o2[r] * inv);
    crow[48] = (bf16_t)(o3[r] * inv);
  }
}

// ---------------------------------------------------------------------------
// Output projection: out = ctx @ Wo + bo (fp32 out)
// ---------------------------------------------------------------------------
__global__ void __launch_bounds__(256) out_proj_kernel(
    const bf16_t* __restrict__ Ctx, const bf16_t* __restrict__ Wot,
    const float* __restrict__ bo, float* __restrict__ Out) {
  const int lane = threadIdx.x & 31;
  const int wave = threadIdx.x >> 5;
  const int half = lane >> 4;
  const int lcol = lane & 15;

  const int tile = blockIdx.x * 8 + wave;           // 0 .. 256*16-1
  const int mrow = tile / (CH / 64);
  const int nt   = tile % (CH / 64);

  const bf16_t* arowp = Ctx + (size_t)(mrow * 16 + lcol) * CH;

  v8f acc[4] = {{}, {}, {}, {}};
  for (int k0 = 0; k0 < CH; k0 += 32) {
    v16bf a = load_a_frag(arowp, k0, half);
#pragma unroll
    for (int j = 0; j < 4; ++j) {
      const int n = nt * 64 + j * 16 + lcol;
      v16bf b = *(const v16bf*)(Wot + (size_t)n * CH + k0 + half * 16);
      acc[j] = wmma_bf16(a, b, acc[j]);
    }
  }
#pragma unroll
  for (int j = 0; j < 4; ++j) {
    const int n = nt * 64 + j * 16 + lcol;
    const float bval = bo[n];
#pragma unroll
    for (int r = 0; r < 8; ++r) {
      const size_t g = (size_t)mrow * 16 + half * 8 + r;
      Out[g * CH + n] = acc[j][r] + bval;
    }
  }
}

// ---------------------------------------------------------------------------
// Host launcher
// ---------------------------------------------------------------------------
extern "C" void kernel_launch(void* const* d_in, const int* in_sizes, int n_in,
                              void* d_out, int out_size, void* d_ws, size_t ws_size,
                              hipStream_t stream) {
  (void)in_sizes; (void)n_in; (void)out_size; (void)ws_size;
  const float* X  = (const float*)d_in[0];
  // d_in[1] = attention_mask (pure causal; applied analytically in-kernel)
  const float* Wq = (const float*)d_in[2];
  const float* bq = (const float*)d_in[3];
  const float* Wk = (const float*)d_in[4];
  const float* bk = (const float*)d_in[5];
  const float* Wv = (const float*)d_in[6];
  const float* bv = (const float*)d_in[7];
  const float* Wo = (const float*)d_in[8];
  const float* bo = (const float*)d_in[9];
  // d_in[10..13] = Wr1/br1/Wr2/br2: dead-end routing MLP, unused by the output.
  float* out = (float*)d_out;

  char* ws = (char*)d_ws;
  const size_t MB = 1024 * 1024;
  bf16_t* Xbf = (bf16_t*)(ws);             // 8 MB   (reused as Ctx later)
  bf16_t* Wqt = (bf16_t*)(ws + 8 * MB);    // 2 MB
  bf16_t* Wkt = (bf16_t*)(ws + 10 * MB);   // 2 MB
  bf16_t* Wvt = (bf16_t*)(ws + 12 * MB);   // 2 MB
  bf16_t* Wot = (bf16_t*)(ws + 14 * MB);   // 2 MB
  bf16_t* Qbf = (bf16_t*)(ws + 16 * MB);   // 8 MB
  bf16_t* Kbf = (bf16_t*)(ws + 24 * MB);   // 8 MB
  bf16_t* Vt  = (bf16_t*)(ws + 32 * MB);   // 8 MB
  bf16_t* Ctx = Xbf;                       // alias: Xbf is dead after QKV GEMM
  // total workspace footprint: 40 MB

  // 1) precision prep
  const int nX = ROWS * CH;
  cvt_bf16_kernel<<<(nX + 255) / 256, 256, 0, stream>>>(X, Xbf, nX);
  dim3 tg(CH / 16, CH / 16), tb(16, 16);
  transpose_bf16_kernel<<<tg, tb, 0, stream>>>(Wq, Wqt);
  transpose_bf16_kernel<<<tg, tb, 0, stream>>>(Wk, Wkt);
  transpose_bf16_kernel<<<tg, tb, 0, stream>>>(Wv, Wvt);
  transpose_bf16_kernel<<<tg, tb, 0, stream>>>(Wo, Wot);

  // 2) fused QKV projection (3 * 256 * 16 wave-tiles, 8 waves/block)
  qkv_gemm_kernel<<<(3 * (ROWS / 16) * (CH / 64)) / 8, 256, 0, stream>>>(
      Xbf, Wqt, Wkt, Wvt, bq, bk, bv, Qbf, Kbf, Vt);

  // 3) flash attention: grid (q-tiles, heads, batch), one wave per block
  attn_kernel<<<dim3(SEQ / 16, HEADS, BATCH), 32, 0, stream>>>(Qbf, Kbf, Vt, Ctx);

  // 4) output projection
  out_proj_kernel<<<((ROWS / 16) * (CH / 64)) / 8, 256, 0, stream>>>(Ctx, Wot, bo, out);
}
